// RBFKernelMappingLayer_78151224918211
// MI455X (gfx1250) — matrix operand order repaired and checked by reading, not compiled, not run on hardware
//
#include <hip/hip_runtime.h>
#include <math.h>

// CDNA5 WMMA vector types
typedef __attribute__((ext_vector_type(16))) __bf16 v16bf;
typedef __attribute__((ext_vector_type(8)))  float  v8f;

#define K_DIM 512
#define N_DIM 2048
#define BM 128
#define BN 64
#define BK 32
#define A_STRIDE (BK + 8)   // 40 bf16 = 80 B row pitch (16B-aligned, conflict-dodging)
#define BT_STRIDE (BK + 8)  // transposed B: n-major rows of BK k-values
#define KSTEPS (K_DIM / BK) // 16

__global__ __launch_bounds__(256)
void rbf_wmma_bf16x3_kernel(const float* __restrict__ X,
                            const float* __restrict__ W,
                            const float* __restrict__ bias,
                            float* __restrict__ out)
{
    // Double-buffered hi/lo bf16 staging (~60 KB of the 320 KB WGP LDS).
    // A is k-major (rows of BK), B is stored TRANSPOSED (n-major rows of BK)
    // so every WMMA fragment is two contiguous ds_load_b128 per lane.
    __shared__ __bf16 As_hi[2][BM][A_STRIDE];
    __shared__ __bf16 As_lo[2][BM][A_STRIDE];
    __shared__ __bf16 Bt_hi[2][BN][BT_STRIDE];
    __shared__ __bf16 Bt_lo[2][BN][BT_STRIDE];

    const int tid  = threadIdx.x;
    const int lane = tid & 31;          // wave32
    const int wave = tid >> 5;          // 0..7
    const int wm   = wave >> 1;         // 0..3 : wave M position (32 rows each)
    const int wn   = wave & 1;          // 0..1 : wave N position (32 cols each)
    const int half = lane >> 4;         // 0/1  : lane half (per ISA VGPR layouts)
    const int l16  = lane & 15;

    const int blockN = blockIdx.x * BN; // N fastest -> X strip reuse hits L2
    const int blockM = blockIdx.y * BM;

    // ---- accumulators seeded with the bias: D = X*W + b ----
    v8f acc[2][2];
#pragma unroll
    for (int mt = 0; mt < 2; ++mt)
#pragma unroll
        for (int nt = 0; nt < 2; ++nt) {
            float bv = bias[blockN + wn * 32 + nt * 16 + l16];
#pragma unroll
            for (int r = 0; r < 8; ++r) acc[mt][nt][r] = bv;
        }

    // ---- global-load prefetch registers (fp32, b128 loads) ----
    float4 aReg[4];  // 128x32 tile: 1024 float4, 4 per thread
    float4 bReg[2];  //  32x64 tile:  512 float4, 2 per thread

    auto loadGlobal = [&](int k0) {
#pragma unroll
        for (int j = 0; j < 4; ++j) {
            int idx = tid + j * 256;
            int row = idx >> 3;            // 8 float4 per 32-float row
            int col = (idx & 7) * 4;
            aReg[j] = *reinterpret_cast<const float4*>(
                &X[(size_t)(blockM + row) * K_DIM + k0 + col]);
        }
#pragma unroll
        for (int j = 0; j < 2; ++j) {
            int idx = tid + j * 256;
            int row = idx >> 4;            // 16 float4 per 64-float row
            int col = (idx & 15) * 4;
            bReg[j] = *reinterpret_cast<const float4*>(
                &W[(size_t)(k0 + row) * N_DIM + blockN + col]);
        }
    };

    auto stageToLds = [&](int buf) {
#pragma unroll
        for (int j = 0; j < 4; ++j) {
            int idx = tid + j * 256;
            int row = idx >> 3;
            int col = (idx & 7) * 4;
            float v[4] = {aReg[j].x, aReg[j].y, aReg[j].z, aReg[j].w};
#pragma unroll
            for (int e = 0; e < 4; ++e) {
                __bf16 hi = (__bf16)v[e];
                __bf16 lo = (__bf16)(v[e] - (float)hi);   // residual term
                As_hi[buf][row][col + e] = hi;
                As_lo[buf][row][col + e] = lo;
            }
        }
#pragma unroll
        for (int j = 0; j < 2; ++j) {
            int idx = tid + j * 256;
            int row = idx >> 4;            // k index within tile
            int col = (idx & 15) * 4;      // n index within tile
            float v[4] = {bReg[j].x, bReg[j].y, bReg[j].z, bReg[j].w};
#pragma unroll
            for (int e = 0; e < 4; ++e) {
                __bf16 hi = (__bf16)v[e];
                __bf16 lo = (__bf16)(v[e] - (float)hi);
                Bt_hi[buf][col + e][row] = hi;             // transposed store
                Bt_lo[buf][col + e][row] = lo;
            }
        }
    };

    loadGlobal(0);

    for (int kt = 0; kt < KSTEPS; ++kt) {
        const int buf = kt & 1;
        stageToLds(buf);
        __syncthreads();                       // stage done before anyone reads buf
        if (kt + 1 < KSTEPS) loadGlobal((kt + 1) * BK);  // overlap next global loads

        // global_prefetch_b8 for the tile two K-steps ahead (keeps L2/HBM busy)
        if (kt + 2 < KSTEPS) {
            const int kp = (kt + 2) * BK;
            __builtin_prefetch(&X[(size_t)(blockM + (tid >> 1)) * K_DIM + kp + (tid & 1) * 16], 0, 3);
            __builtin_prefetch(&W[(size_t)(kp + (tid >> 3)) * N_DIM + blockN + (tid & 7) * 8], 0, 3);
        }

        // ---- load WMMA fragments per documented CDNA5 VGPR striping ----
        // A 16x32 bf16: elems 0..7 -> K = half*8 + j ; elems 8..15 -> K = 16 + half*8 + j
        v16bf aHi[2], aLo[2];
#pragma unroll
        for (int mt = 0; mt < 2; ++mt) {
            const int m = wm * 32 + mt * 16 + l16;
#pragma unroll
            for (int j = 0; j < 8; ++j) {
                aHi[mt][j]     = As_hi[buf][m][half * 8 + j];
                aHi[mt][j + 8] = As_hi[buf][m][16 + half * 8 + j];
                aLo[mt][j]     = As_lo[buf][m][half * 8 + j];
                aLo[mt][j + 8] = As_lo[buf][m][16 + half * 8 + j];
            }
        }
        // B 32x16 bf16: elems 0..7 -> K = half*8 + j ; elems 8..15 -> K = 16 + half*8 + j; N = lane&15.
        // B lives transposed in LDS, so each group of 8 k-values per lane is contiguous.
        v16bf bHi[2], bLo[2];
#pragma unroll
        for (int nt = 0; nt < 2; ++nt) {
            const int n  = wn * 32 + nt * 16 + l16;
            const int kb = half * 8;
#pragma unroll
            for (int j = 0; j < 8; ++j) {
                bHi[nt][j]     = Bt_hi[buf][n][kb + j];
                bHi[nt][j + 8] = Bt_hi[buf][n][16 + kb + j];
                bLo[nt][j]     = Bt_lo[buf][n][kb + j];
                bLo[nt][j + 8] = Bt_lo[buf][n][16 + kb + j];
            }
        }

        // ---- bf16x3 full-precision GEMM: hi*hi + hi*lo + lo*hi ----
#pragma unroll
        for (int mt = 0; mt < 2; ++mt)
#pragma unroll
            for (int nt = 0; nt < 2; ++nt) {
                acc[mt][nt] = __builtin_amdgcn_wmma_f32_16x16x32_bf16(
                    false, aHi[mt], false, bHi[nt], (short)0, acc[mt][nt], false, false);
                acc[mt][nt] = __builtin_amdgcn_wmma_f32_16x16x32_bf16(
                    false, aHi[mt], false, bLo[nt], (short)0, acc[mt][nt], false, false);
                acc[mt][nt] = __builtin_amdgcn_wmma_f32_16x16x32_bf16(
                    false, aLo[mt], false, bHi[nt], (short)0, acc[mt][nt], false, false);
            }
        // Double buffer: next iteration stages into buf^1, so one barrier/iter suffices.
    }

    // ---- epilogue: cos(proj) * sqrt(2/2048); sqrt(2/2048) = 1/32 exactly ----
    // Non-temporal stores: the 512 MB output stream must NOT evict the 128 MB
    // X working set from the 192 MB L2 (the whole A-reuse scheme depends on it).
    const float scale = 0.03125f;
#pragma unroll
    for (int mt = 0; mt < 2; ++mt)
#pragma unroll
        for (int nt = 0; nt < 2; ++nt) {
            const int n = blockN + wn * 32 + nt * 16 + l16;
#pragma unroll
            for (int r = 0; r < 8; ++r) {
                // C/D layout: VGPR r -> M = r (lanes 0-15) or r+8 (lanes 16-31)
                const int m = blockM + wm * 32 + mt * 16 + half * 8 + r;
                __builtin_nontemporal_store(cosf(acc[mt][nt][r]) * scale,
                                            &out[(size_t)m * N_DIM + n]);
            }
        }
}

extern "C" void kernel_launch(void* const* d_in, const int* in_sizes, int n_in,
                              void* d_out, int out_size, void* d_ws, size_t ws_size,
                              hipStream_t stream) {
    const float* X    = (const float*)d_in[0];  // [N_ROWS, 512]
    const float* W    = (const float*)d_in[1];  // [512, 2048]
    const float* bias = (const float*)d_in[2];  // [2048]
    float* out        = (float*)d_out;          // [N_ROWS, 2048]

    const int nrows = in_sizes[0] / K_DIM;      // 65536
    dim3 grid(N_DIM / BN, nrows / BM);          // (32, 512); N fastest for L2 reuse of X
    dim3 block(256);                            // 8 wave32s
    rbf_wmma_bf16x3_kernel<<<grid, block, 0, stream>>>(X, W, bias, out);
}
